// CombinedLoss_22385369547079
// MI455X (gfx1250) — compile-verified, matrix operand order checked by reading, not run on hardware
//
#include <hip/hip_runtime.h>
#include <hip/hip_bf16.h>
#include <math.h>

// ---------------- CDNA5 (gfx1250) wave32 WMMA types ----------------
typedef __attribute__((ext_vector_type(16))) __bf16 v16bf;
typedef __attribute__((ext_vector_type(8)))  float  v8f;

#define B_IMG 16
#define HW    1089            // 33*33 pixels per image
#define NPAD  1104            // 69 tiles of 16 rows
#define DIM   512
#define NMASK 6
#define NLBL  7               // labels 0..6
#define NTILE 69
#define INVT  (1.0f/0.07f)    // 1/temperature

// ---- workspace layout (bytes) ----
#define OFF_LABELS   0u                       // B*NPAD int
#define OFF_RN2      70656u                   // B*NPAD float  (self dot per pixel)
#define OFF_CNT      141312u                  // B*8 float
#define OFF_Q        141824u                  // B*8 float (per-label sum of ||e||^2)
#define OFF_ACC      142336u                  // B*2 float (supcon numerator, anchor count)
#define OFF_CG       142464u                  // B*64 float (centroid gram + total-sum dot)
#define OFF_SUMS     146560u                  // B*7*512 float (per-label feature sums)
#define OFF_FBF      375936u                  // B*NPAD*512 bf16 (32B aligned)
// total ~18.5 MB

// ---------------- init: zero pads / accumulators ----------------
__global__ void k_init(int* labels, float* rn2, float* q, float* accum, __bf16* fbf) {
  int idx = blockIdx.x * blockDim.x + threadIdx.x;
  const int padElems = B_IMG * (NPAD - HW) * DIM;      // 122880
  if (idx < padElems) {
    int b   = idx / ((NPAD - HW) * DIM);
    int rem = idx % ((NPAD - HW) * DIM);
    int row = HW + rem / DIM;
    int d   = rem % DIM;
    fbf[((size_t)b * NPAD + row) * DIM + d] = (__bf16)0.0f;
  }
  if (idx < B_IMG * (NPAD - HW)) {
    int b = idx / (NPAD - HW), row = HW + idx % (NPAD - HW);
    labels[b * NPAD + row] = -1;                       // never matches a real label
    rn2[b * NPAD + row]    = 0.f;
  }
  if (idx < B_IMG * 8) q[idx]     = 0.f;
  if (idx < B_IMG * 2) accum[idx] = 0.f;
}

// ---------------- labels from masks (one block per image) ----------------
__global__ void k_labels(const float* __restrict__ masks, int* labels, float* cnt) {
  int b = blockIdx.x, tid = threadIdx.x;
  __shared__ float red[256];
  __shared__ int   active[NMASK];
  __shared__ int   cshare[NLBL];
  const float* mb = masks + (size_t)b * NMASK * HW;
  for (int m = 0; m < NMASK; m++) {
    float s = 0.f;
    for (int n = tid; n < HW; n += 256) s += mb[m * HW + n];
    red[tid] = s; __syncthreads();
    for (int o = 128; o > 0; o >>= 1) { if (tid < o) red[tid] += red[tid + o]; __syncthreads(); }
    if (tid == 0) active[m] = (red[0] > 0.f);
    __syncthreads();
  }
  if (tid < NLBL) cshare[tid] = 0;
  __syncthreads();
  for (int n = tid; n < HW; n += 256) {
    int lbl = 0;
    #pragma unroll
    for (int m = 0; m < NMASK; m++)
      if (active[m] && mb[m * HW + n] > 0.5f) lbl = m + 1;   // later channels win == max
    labels[b * NPAD + n] = lbl;
    atomicAdd(&cshare[lbl], 1);
  }
  __syncthreads();
  if (tid < NLBL) cnt[b * 8 + tid] = (float)cshare[tid];
}

// ---------------- fp32 -> bf16 + per-pixel self-dot (one wave32 per pixel) ----------------
__global__ void k_convert(const float* __restrict__ emb, __bf16* fbf, float* rn2) {
  int wid  = (blockIdx.x * blockDim.x + threadIdx.x) >> 5;
  int lane = threadIdx.x & 31;
  if (wid >= B_IMG * HW) return;
  int b = wid / HW, n = wid % HW;
  const float* src = emb + ((size_t)b * HW + n) * DIM + lane * 16;
  __bf16*      dst = fbf + ((size_t)b * NPAD + n) * DIM + lane * 16;
  float ss = 0.f;
  v16bf out;
  #pragma unroll
  for (int i = 0; i < 16; i++) { float x = src[i]; ss += x * x; out[i] = (__bf16)x; }
  *(v16bf*)dst = out;
  #pragma unroll
  for (int m = 16; m > 0; m >>= 1) ss += __shfl_xor(ss, m, 32);
  if (lane == 0) rn2[b * NPAD + n] = ss;
}

// ---------------- per-label sums + sum of squared norms ----------------
__global__ void k_stats(const float* __restrict__ emb, const int* __restrict__ labels,
                        float* sums, float* q) {
  int b = blockIdx.x >> 1;
  int d = (blockIdx.x & 1) * 256 + threadIdx.x;
  const float* eb = emb + (size_t)b * HW * DIM + d;
  const int*   lb = labels + b * NPAD;
  float s[NLBL] = {0}, qp[NLBL] = {0};
  for (int n = 0; n < HW; n++) {
    int   l = lb[n];
    float x = eb[(size_t)n * DIM];
    #pragma unroll
    for (int k = 0; k < NLBL; k++) { bool c = (l == k); s[k] += c ? x : 0.f; qp[k] += c ? x * x : 0.f; }
  }
  #pragma unroll
  for (int k = 0; k < NLBL; k++) {
    sums[((size_t)b * NLBL + k) * DIM + d] = s[k];
    atomicAdd(&q[b * 8 + k], qp[k]);
  }
}

// ---------------- SupCon: tiled Gram via v_wmma_f32_16x16x32_bf16 ----------------
// One wave owns a 16-anchor tile; A fragments (16 K-chunks) live in VGPRs across
// all 69 j-tiles. C layout: lane = column j, vgpr r = row r (lanes 0-15) / r+8 (16-31).
__global__ void __launch_bounds__(256, 1)
k_supcon(const __bf16* __restrict__ fbf, const int* __restrict__ labels,
         const float* __restrict__ rn2, float* accum) {
  int wid  = blockIdx.x * (blockDim.x >> 5) + (threadIdx.x >> 5);
  int lane = threadIdx.x & 31;
  if (wid >= B_IMG * NTILE) return;       // wave-uniform: EXEC stays all-ones for WMMA
  int b = wid / NTILE, it = wid % NTILE;
  int ibase   = it * 16;
  const __bf16* fb = fbf + (size_t)b * NPAD * DIM;
  const int*    lb = labels + b * NPAD;
  const float*  rb = rn2 + b * NPAD;
  int halfOff = (lane >> 4) << 4;         // K offset 0 / 16 (upper half-wave)
  int col     = lane & 15;
  int rowsel  = (lane >> 4) << 3;         // C rows 0..7 / 8..15 per half

  v16bf afrag[16];
  const __bf16* arow = fb + (size_t)(ibase + col) * DIM + halfOff;
  #pragma unroll
  for (int c = 0; c < 16; c++) afrag[c] = *(const v16bf*)(arow + c * 32);

  int li[8]; float sd[8];
  #pragma unroll
  for (int r = 0; r < 8; r++) { int m = ibase + rowsel + r; li[r] = lb[m]; sd[r] = rb[m]; }

  float esum[8] = {0}, plog[8] = {0}, pcnt[8] = {0};
  for (int jt = 0; jt < NTILE; jt++) {
    int jbase = jt * 16;
    int jg    = jbase + col;
    int lj    = lb[jg];
    const __bf16* brow = fb + (size_t)jg * DIM + halfOff;
    v8f acc = {};
    #pragma unroll
    for (int c = 0; c < 16; c++) {
      v16bf bfrag = *(const v16bf*)(brow + c * 32);
      acc = __builtin_amdgcn_wmma_f32_16x16x32_bf16(false, afrag[c], false, bfrag,
                                                    (short)0, acc, false, false);
    }
    #pragma unroll
    for (int r = 0; r < 8; r++) {
      int  m     = ibase + rowsel + r;
      bool valid = (m < HW) && (jg < HW) && (m != jg);
      float lc   = (acc[r] - sd[r]) * INVT;          // center by row max (= self dot)
      esum[r] += valid ? __expf(lc) : 0.f;
      bool same = valid && (li[r] == lj);
      plog[r] += same ? lc : 0.f;
      pcnt[r] += same ? 1.f : 0.f;
    }
  }
  // butterfly reduce the two 16-lane halves (matches C-matrix row split)
  #pragma unroll
  for (int r = 0; r < 8; r++) {
    #pragma unroll
    for (int msk = 8; msk > 0; msk >>= 1) {
      esum[r] += __shfl_xor(esum[r], msk, 32);
      plog[r] += __shfl_xor(plog[r], msk, 32);
      pcnt[r] += __shfl_xor(pcnt[r], msk, 32);
    }
  }
  float lsum = 0.f, lcnt = 0.f;
  #pragma unroll
  for (int r = 0; r < 8; r++) {
    int m = ibase + rowsel + r;
    if (m < HW && pcnt[r] > 0.f) {
      float mlp = (plog[r] - pcnt[r] * __logf(esum[r] + 1e-6f)) / pcnt[r];
      lsum += -mlp; lcnt += 1.f;
    }
  }
  lsum += __shfl_xor(lsum, 16, 32);
  lcnt += __shfl_xor(lcnt, 16, 32);
  if (lane == 0) {
    atomicAdd(&accum[b * 2 + 0], lsum);
    atomicAdd(&accum[b * 2 + 1], lcnt);
  }
}

// ---------------- 7x7 centroid gram + total-sum self-dot ----------------
__global__ void k_centroid(const float* __restrict__ sums, const float* __restrict__ cnt,
                           float* cg) {
  int b = blockIdx.x, t = threadIdx.x;
  const float* sb = sums + (size_t)b * NLBL * DIM;
  if (t < 49) {
    int i = t / 7, j = t % 7;
    float inv_i = 1.f / fmaxf(cnt[b * 8 + i], 1.f);
    float inv_j = 1.f / fmaxf(cnt[b * 8 + j], 1.f);
    float dot = 0.f;
    for (int d = 0; d < DIM; d++) dot += sb[i * DIM + d] * sb[j * DIM + d];
    cg[b * 64 + t] = dot * inv_i * inv_j;
  } else if (t == 49) {
    float ss = 0.f;
    for (int d = 0; d < DIM; d++) {
      float tot = 0.f;
      #pragma unroll
      for (int l = 0; l < NLBL; l++) tot += sb[l * DIM + d];
      ss += tot * tot;
    }
    cg[b * 64 + 49] = ss;
  }
}

// ---------------- final scalar combine ----------------
__global__ void k_final(const float* __restrict__ cnt, const float* __restrict__ q,
                        const float* __restrict__ accum, const float* __restrict__ cg,
                        const float* __restrict__ forged, float* out) {
  int b = threadIdx.x;
  __shared__ float sv[B_IMG][6];
  if (b < B_IMG) {
    float c[7], qq[7], g[49];
    for (int l = 0; l < 7; l++) { c[l] = cnt[b * 8 + l]; qq[l] = q[b * 8 + l]; }
    for (int t = 0; t < 49; t++) g[t] = cg[b * 64 + t];
    float ss = cg[b * 64 + 49];
    // supcon
    float na  = accum[b * 2 + 1];
    float sc  = accum[b * 2 + 0] / fmaxf(na, 1.f);
    float scv = (na > 0.f) ? 1.f : 0.f;
    // separation
    float sep = 0.f, np = 0.f, npres = 0.f;
    for (int l = 0; l < 7; l++) npres += (c[l] > 0.f) ? 1.f : 0.f;
    for (int i = 0; i < 7; i++)
      for (int j = i + 1; j < 7; j++)
        if (c[i] > 0.f && c[j] > 0.f) {
          float sq = g[i * 7 + i] + g[j * 7 + j] - 2.f * g[i * 7 + j];
          float d  = sqrtf(fmaxf(sq, 0.f));
          sep += fmaxf(2.0f - d, 0.f);
          np  += 1.f;
        }
    sep = sep / fmaxf(np, 1.f);
    float spv = (npres >= 2.f) ? 1.f : 0.f;
    // uniformity
    float il = 0.f, nl = 0.f;
    for (int l = 0; l < 7; l++) {
      float var = qq[l] / fmaxf(c[l], 1.f) - g[l * 7 + l];
      if (c[l] >= 2.f && var > 0.1f) { il += var - 0.1f; nl += 1.f; }
    }
    float uf = il / fmaxf(nl, 1.f);
    float ufv = (nl > 0.f) ? 1.f : 0.f;
    float qtot = 0.f; for (int l = 0; l < 7; l++) qtot += qq[l];
    float Nf = (float)HW;
    float var_all = qtot / Nf - ss / (Nf * Nf);
    float uu  = (var_all > 0.1f) ? (var_all - 0.1f) : 0.f;
    float uuv = (var_all > 0.1f) ? 1.f : 0.f;
    bool isf = forged[b] >= 0.5f;
    float un  = isf ? uf  : uu;
    float unv = isf ? ufv : uuv;
    sv[b][0] = sc * scv;  sv[b][1] = scv;
    sv[b][2] = sep * spv; sv[b][3] = spv;
    sv[b][4] = un * unv;  sv[b][5] = unv;
  }
  __syncthreads();
  if (threadIdx.x == 0) {
    float a[6] = {0, 0, 0, 0, 0, 0};
    for (int i = 0; i < B_IMG; i++)
      for (int k = 0; k < 6; k++) a[k] += sv[i][k];
    float sup = (a[1] > 0.f) ? a[0] / fmaxf(a[1], 1.f) : 0.f;
    float sep = (a[3] > 0.f) ? a[2] / fmaxf(a[3], 1.f) : 0.f;
    float uni = (a[5] > 0.f) ? a[4] / fmaxf(a[5], 1.f) : 0.f;
    out[0] = 1.0f * sup + 0.5f * sep + 0.5f * uni;
  }
}

extern "C" void kernel_launch(void* const* d_in, const int* in_sizes, int n_in,
                              void* d_out, int out_size, void* d_ws, size_t ws_size,
                              hipStream_t stream) {
  const float* emb    = (const float*)d_in[0];   // (16,33,33,512) f32, unit-normalized
  const float* masks  = (const float*)d_in[1];   // (16,6,33,33) f32
  const float* forged = (const float*)d_in[2];   // (16,) f32
  float* out = (float*)d_out;

  char* ws = (char*)d_ws;
  int*    labels = (int*)   (ws + OFF_LABELS);
  float*  rn2    = (float*) (ws + OFF_RN2);
  float*  cnt    = (float*) (ws + OFF_CNT);
  float*  q      = (float*) (ws + OFF_Q);
  float*  accum  = (float*) (ws + OFF_ACC);
  float*  cg     = (float*) (ws + OFF_CG);
  float*  sums   = (float*) (ws + OFF_SUMS);
  __bf16* fbf    = (__bf16*)(ws + OFF_FBF);

  k_init    <<<480, 256, 0, stream>>>(labels, rn2, q, accum, fbf);
  k_labels  <<<B_IMG, 256, 0, stream>>>(masks, labels, cnt);
  k_convert <<<(B_IMG * HW + 7) / 8, 256, 0, stream>>>(emb, fbf, rn2);
  k_stats   <<<B_IMG * 2, 256, 0, stream>>>(emb, labels, sums, q);
  k_supcon  <<<(B_IMG * NTILE + 7) / 8, 256, 0, stream>>>(fbf, labels, rn2, accum);
  k_centroid<<<B_IMG, 64, 0, stream>>>(sums, cnt, cg);
  k_final   <<<1, 32, 0, stream>>>(cnt, q, accum, cg, forged, out);
}
